// GIN_11622181503634
// MI455X (gfx1250) — compile-verified, matrix operand order checked by reading, not run on hardware
//
#include <hip/hip_runtime.h>

#define N_NODES 100000
#define N_EDGES 1200000
#define DIM 64
#define HID 64
#define TGT 32
#define NLAYERS 3
#define NGRAPH 500

#define STRIPS 10          // 16-row strips per block (6250 strips = 625 blocks exactly)
#define MLP_BLOCK (32 * STRIPS)
#define TPAD 72            // LDS row stride in halves (64 + 8 pad; spreads banks, keeps 16B align)

typedef _Float16 v16h __attribute__((ext_vector_type(16)));
typedef _Float16 v8h  __attribute__((ext_vector_type(8)));
typedef float    v8f  __attribute__((ext_vector_type(8)));

__device__ __forceinline__ void fatomic_add(float* p, float v) {
    unsafeAtomicAdd(p, v);   // -> global_atomic_add_f32 (L2 atomic units)
}

__device__ __forceinline__ v16h ld_frag(const _Float16* p) {
    // two aligned 16B LDS loads -> one WMMA operand (two contiguous 8-half K-runs)
    v8h lo = *(const v8h*)(p);
    v8h hi = *(const v8h*)(p + 16);
    return __builtin_shufflevector(lo, hi, 0, 1, 2, 3, 4, 5, 6, 7,
                                   8, 9, 10, 11, 12, 13, 14, 15);
}

__global__ void k_zero(float* p, int n) {
    int i = blockIdx.x * blockDim.x + threadIdx.x;
    if (i < n) p[i] = 0.0f;
}

__global__ void k_copy(const float* __restrict__ x, float* __restrict__ y, int n) {
    int i = blockIdx.x * blockDim.x + threadIdx.x;
    if (i < n) y[i] = x[i];
}

__global__ void k_counts(const int* __restrict__ gid, float* counts) {
    int i = blockIdx.x * blockDim.x + threadIdx.x;
    if (i < N_NODES) fatomic_add(&counts[gid[i]], 1.0f);
}

// agg = (1 + eps[l]) * feat   (GINConv self term)
__global__ void k_init_agg(const float* __restrict__ feat, const float* __restrict__ eps,
                           int l, float* __restrict__ agg) {
    int i = blockIdx.x * blockDim.x + threadIdx.x;
    if (i < N_NODES * HID) agg[i] = (1.0f + eps[l]) * feat[i];
}

// Edge-parallel scatter-add: thread = (edge, 4-channel chunk).
// feat table (25.6MB) is resident in the 192MB L2; atomics hit L2 atomic units.
__global__ void k_scatter(const float* __restrict__ feat, const int* __restrict__ src,
                          const int* __restrict__ dst, float* __restrict__ agg) {
    int idx = blockIdx.x * blockDim.x + threadIdx.x;
    if (idx >= N_EDGES * (HID / 4)) return;
    int e = idx >> 4;
    int c = (idx & 15) << 2;
    int s = src[e];
    int d = dst[e];
    const float4 v = *(const float4*)(feat + (size_t)s * HID + c);
    float* o = agg + (size_t)d * HID + c;
    fatomic_add(o + 0, v.x);
    fatomic_add(o + 1, v.y);
    fatomic_add(o + 2, v.z);
    fatomic_add(o + 3, v.w);
}

// ---------------------------------------------------------------------------
// Fused 2-layer MLP: out[N x NC2] = relu(A[N x 64] @ W1 + bs1) @ W2 + bs2
// - W1/W2 staged once per block into LDS, transposed + converted to f16, so
//   every B fragment is two ds_load_b128 (per-lane contiguous 8-half K-runs).
// - One wave per 16-row strip; stage-1 hidden tile lives in 4 f32 WMMA
//   accumulators, goes through LDS as f16, reloads in WMMA-A layout.
// WMMA f16 lane layout (ISA 7.12.2): lane L (mn=L&15, half=L>>4), elem e:
//   k = (e>>3)*16 + half*8 + (e&7);  C/D: VGPR r -> row r+8*half, col L&15.
// ---------------------------------------------------------------------------
template <int NC2>
__global__ void k_mlp_wmma(const float* __restrict__ A,
                           const float* __restrict__ W1, const float* __restrict__ bs1,
                           const float* __restrict__ W2, const float* __restrict__ bs2,
                           float* __restrict__ out) {
    __shared__ alignas(16) _Float16 tile[STRIPS][16 * TPAD];
    __shared__ alignas(16) _Float16 w1t[64 * TPAD];          // w1t[n][k] = W1[k][n]
    __shared__ alignas(16) _Float16 w2t[NC2 * TPAD];         // w2t[n][k] = W2[k][n]

    // ---- cooperative weight staging: coalesced global reads, f32->f16 once ----
    for (int idx = threadIdx.x; idx < 64 * 64; idx += MLP_BLOCK) {
        int k = idx >> 6, n = idx & 63;
        w1t[n * TPAD + k] = (_Float16)W1[idx];
    }
    for (int idx = threadIdx.x; idx < 64 * NC2; idx += MLP_BLOCK) {
        int k = idx / NC2, n = idx % NC2;                    // NC2 is a power of two
        w2t[n * TPAD + k] = (_Float16)W2[idx];
    }
    __syncthreads();

    const int wave = threadIdx.x >> 5;
    const int lane = threadIdx.x & 31;
    const int strip = blockIdx.x * STRIPS + wave;
    const int mn = lane & 15;
    const int half = lane >> 4;
    const int row0 = strip * 16;

    // ---- Stage 1: load A strip once (registers), 4 col tiles of W1 ----
    const float* arow = A + (size_t)(row0 + mn) * 64 + half * 8;
    v16h a0, a1;
#pragma unroll
    for (int e = 0; e < 8; ++e) {
        a0[e]     = (_Float16)arow[e];
        a0[e + 8] = (_Float16)arow[16 + e];
        a1[e]     = (_Float16)arow[32 + e];
        a1[e + 8] = (_Float16)arow[48 + e];
    }

    v8f acc[4];
#pragma unroll
    for (int ct = 0; ct < 4; ++ct) {
        const _Float16* wrow = w1t + (ct * 16 + mn) * TPAD + half * 8;
        v16h bf0 = ld_frag(wrow);        // K 0..31 runs
        v16h bf1 = ld_frag(wrow + 32);   // K 32..63 runs
        v8f c = {};
        c = __builtin_amdgcn_wmma_f32_16x16x32_f16(false, a0, false, bf0, (short)0, c, false, false);
        c = __builtin_amdgcn_wmma_f32_16x16x32_f16(false, a1, false, bf1, (short)0, c, false, false);
        float bc = bs1[ct * 16 + mn];
#pragma unroll
        for (int r = 0; r < 8; ++r) c[r] = fmaxf(c[r] + bc, 0.0f);
        acc[ct] = c;
    }

    // ---- Hidden tile -> LDS as f16 (row-major, stride TPAD) ----
    _Float16* t = &tile[wave][0];
#pragma unroll
    for (int ct = 0; ct < 4; ++ct)
#pragma unroll
        for (int r = 0; r < 8; ++r)
            t[(r + 8 * half) * TPAD + ct * 16 + mn] = (_Float16)acc[ct][r];

    __syncthreads();   // non-divergent; orders LDS store->load (waves use own region)

    // ---- Stage 2: reload hidden tile in A layout, GEMM vs W2 (from LDS) ----
    const _Float16* trow = t + mn * TPAD + half * 8;
    v16h h0 = ld_frag(trow);
    v16h h1 = ld_frag(trow + 32);

#pragma unroll
    for (int ct = 0; ct < NC2 / 16; ++ct) {
        const _Float16* wrow = w2t + (ct * 16 + mn) * TPAD + half * 8;
        v16h bf0 = ld_frag(wrow);
        v16h bf1 = ld_frag(wrow + 32);
        v8f c = {};
        c = __builtin_amdgcn_wmma_f32_16x16x32_f16(false, h0, false, bf0, (short)0, c, false, false);
        c = __builtin_amdgcn_wmma_f32_16x16x32_f16(false, h1, false, bf1, (short)0, c, false, false);
        float bc = bs2[ct * 16 + mn];
#pragma unroll
        for (int r = 0; r < 8; ++r)
            out[(size_t)(row0 + r + 8 * half) * NC2 + ct * 16 + mn] = c[r] + bc;
    }
}

// Single-pass GraphNorm statistics: sum and sum-of-squares
__global__ void k_seg_acc2(const float* __restrict__ h, const int* __restrict__ gid,
                           float* __restrict__ sum, float* __restrict__ sumsq) {
    int i = blockIdx.x * blockDim.x + threadIdx.x;
    if (i >= N_NODES * HID) return;
    int node = i >> 6;
    int c = i & 63;
    int g = gid[node];
    float v = h[i];
    fatomic_add(&sum[g * HID + c], v);
    fatomic_add(&sumsq[g * HID + c], v * v);
}

// sum -> mean, sumsq -> inv_std, using E[(h-m*s)^2] = E[h^2] - m^2*s*(2-s)
__global__ void k_gn_fin(float* __restrict__ sum, float* __restrict__ sumsq,
                         const float* __restrict__ counts, const float* __restrict__ gscale,
                         int l) {
    int i = blockIdx.x * blockDim.x + threadIdx.x;
    if (i >= NGRAPH * HID) return;
    float cnt = fmaxf(counts[i >> 6], 1.0f);
    float m = sum[i] / cnt;
    float sc = gscale[l * HID + (i & 63)];
    float var = sumsq[i] / cnt - m * m * sc * (2.0f - sc);
    sum[i] = m;
    sumsq[i] = rsqrtf(fmaxf(var, 0.0f) + 1e-8f);
}

__global__ void k_norm(const float* __restrict__ h, const int* __restrict__ gid,
                       const float* __restrict__ mean, const float* __restrict__ invstd,
                       const float* __restrict__ gw, const float* __restrict__ gb,
                       const float* __restrict__ gscale, int l, float* __restrict__ feat) {
    int i = blockIdx.x * blockDim.x + threadIdx.x;
    if (i >= N_NODES * HID) return;
    int node = i >> 6;
    int c = i & 63;
    int g = gid[node];
    float sub = h[i] - mean[g * HID + c] * gscale[l * HID + c];
    float v = gw[l * HID + c] * sub * invstd[g * HID + c] + gb[l * HID + c];
    feat[i] = fmaxf(v, 0.0f);
}

__global__ void k_pool_acc(const float* __restrict__ z, const int* __restrict__ gid,
                           float* __restrict__ out, int l) {
    int i = blockIdx.x * blockDim.x + threadIdx.x;
    if (i >= N_NODES * TGT) return;
    int node = i >> 5;
    int t = i & 31;
    fatomic_add(&out[gid[node] * (NLAYERS * TGT) + l * TGT + t], z[i]);
}

__global__ void k_out_fin(float* out, const float* __restrict__ counts) {
    int i = blockIdx.x * blockDim.x + threadIdx.x;
    if (i < NGRAPH * NLAYERS * TGT) out[i] /= fmaxf(counts[i / (NLAYERS * TGT)], 1.0f);
}

static inline int cdiv(int a, int b) { return (a + b - 1) / b; }

extern "C" void kernel_launch(void* const* d_in, const int* in_sizes, int n_in,
                              void* d_out, int out_size, void* d_ws, size_t ws_size,
                              hipStream_t stream) {
    const float* x   = (const float*)d_in[0];
    const int*   src = (const int*)d_in[1];
    const int*   dst = (const int*)d_in[2];
    const int*   gid = (const int*)d_in[3];
    const float* eps = (const float*)d_in[4];
    const float* cw1 = (const float*)d_in[5];
    const float* cb1 = (const float*)d_in[6];
    const float* cw2 = (const float*)d_in[7];
    const float* cb2 = (const float*)d_in[8];
    const float* gnw = (const float*)d_in[9];
    const float* gnb = (const float*)d_in[10];
    const float* gns = (const float*)d_in[11];
    const float* pw1 = (const float*)d_in[12];
    const float* pb1 = (const float*)d_in[13];
    const float* pw2 = (const float*)d_in[14];
    const float* pb2 = (const float*)d_in[15];
    float* out = (float*)d_out;

    // Workspace carve (~64 MB f32)
    float* feat   = (float*)d_ws;                   // [N,H] running features
    float* bufA   = feat + (size_t)N_NODES * HID;   // agg -> h (conv MLP in-place)
    float* bufC   = bufA + (size_t)N_NODES * HID;   // proj output [N,T]
    float* mean   = bufC + (size_t)N_NODES * TGT;   // sum -> mean
    float* var    = mean + NGRAPH * HID;            // sumsq -> inv_std
    float* counts = var  + NGRAPH * HID;

    const int B = 256;
    const int mlp_grid = N_NODES / 16 / STRIPS;     // 625, exact

    k_zero<<<cdiv(NGRAPH, B), B, 0, stream>>>(counts, NGRAPH);
    k_zero<<<cdiv(NGRAPH * NLAYERS * TGT, B), B, 0, stream>>>(out, NGRAPH * NLAYERS * TGT);
    k_counts<<<cdiv(N_NODES, B), B, 0, stream>>>(gid, counts);
    k_copy<<<cdiv(N_NODES * DIM, B), B, 0, stream>>>(x, feat, N_NODES * DIM);

    for (int l = 0; l < NLAYERS; ++l) {
        // GINConv aggregation
        k_init_agg<<<cdiv(N_NODES * HID, B), B, 0, stream>>>(feat, eps, l, bufA);
        k_scatter<<<cdiv(N_EDGES * (HID / 4), B), B, 0, stream>>>(feat, src, dst, bufA);
        // conv MLP fused (in-place: each wave reads its 16 rows before writing them)
        k_mlp_wmma<HID><<<mlp_grid, MLP_BLOCK, 0, stream>>>(
            bufA, cw1 + (size_t)l * DIM * HID, cb1 + l * HID,
            cw2 + (size_t)l * HID * HID, cb2 + l * HID, bufA);
        // GraphNorm (single statistics pass)
        k_zero<<<cdiv(NGRAPH * HID, B), B, 0, stream>>>(mean, NGRAPH * HID);
        k_zero<<<cdiv(NGRAPH * HID, B), B, 0, stream>>>(var, NGRAPH * HID);
        k_seg_acc2<<<cdiv(N_NODES * HID, B), B, 0, stream>>>(bufA, gid, mean, var);
        k_gn_fin<<<cdiv(NGRAPH * HID, B), B, 0, stream>>>(mean, var, counts, gns, l);
        k_norm<<<cdiv(N_NODES * HID, B), B, 0, stream>>>(bufA, gid, mean, var, gnw, gnb, gns, l, feat);
        // projection MLP fused + pooled
        k_mlp_wmma<TGT><<<mlp_grid, MLP_BLOCK, 0, stream>>>(
            feat, pw1 + (size_t)l * HID * HID, pb1 + l * HID,
            pw2 + (size_t)l * HID * TGT, pb2 + l * TGT, bufC);
        k_pool_acc<<<cdiv(N_NODES * TGT, B), B, 0, stream>>>(bufC, gid, out, l);
    }
    k_out_fin<<<cdiv(NGRAPH * NLAYERS * TGT, B), B, 0, stream>>>(out, counts);
}
